// SFM_Model_73916387164717
// MI455X (gfx1250) — matrix operand order repaired and checked by reading, not compiled, not run on hardware
//
#include <hip/hip_runtime.h>
#include <hip/hip_bf16.h>

// SFM RNN for MI455X (gfx1250).
// B=256, D=128, T=64, H=256, F=16. Batch rows independent -> 16 blocks,
// each owns a 16-row band and runs the whole 64-step recurrence.
// Recurrent GEMM: [x_t|h](16x384) @ Wcat(384x1040) via v_wmma_f32_16x16x32_bf16.
// A fragments staged in LDS; B panels prefetched one tile ahead (L0-warm).
// S_re/S_im state: v2f16-packed freq pairs, updated with packed-f16 VALU
// (v_pk_mul/v_pk_fma) and reduced with v_dot2_f32_f16 into f32.

#define BB   256
#define DD   128
#define TT   64
#define HH   256
#define FF   16
#define NTOT 1040             // 4*H + F = 65 tiles of 16
#define NT   65
#define KT   12               // 384 / 32
#define KTX  4                // x part:  kt 0..3   (K 0..127)
#define KTH  8                // h part:  kt 4..11  (K 128..383)
#define PRE_STRIDE 1056       // 1040 + 16 pad
#define H_STRIDE   260
#define NTHR 512              // 16 waves per block

typedef __attribute__((ext_vector_type(16))) __bf16    v16bf;
typedef __attribute__((ext_vector_type(8)))  float     v8f;
typedef __attribute__((ext_vector_type(2)))  _Float16  h2f;

// ws layout
#define BPACK_ELEMS   (NT * KT * 512)            // 399,360 u16
#define XFRAG_OFF_U16 (832 * 1024)               // u16 offset, 1.6MB past base
#define XFRAG_ELEMS   (16 * TT * KTX * 512)      // 2,097,152 u16 (4 MB)

static __device__ __forceinline__ unsigned short f2bf_u16(float f) {
    unsigned u = __float_as_uint(f);
    return (unsigned short)((u + 0x7FFFu + ((u >> 16) & 1u)) >> 16);  // RNE
}

static __device__ __forceinline__ float hsig(float x) {
    return fminf(fmaxf(x * (1.0f / 6.0f) + 0.5f, 0.0f), 1.0f);
}

static __device__ __forceinline__ h2f splat2(float x) {
    return (h2f){(_Float16)x, (_Float16)x};
}

// A-fragment K index for element e, lane l (16-bit A 16x32 layout)
static __device__ __forceinline__ int a_koff(int e, int lane_f) {
    return e + ((e >= 8) ? 8 : 0) + ((lane_f >= 16) ? 8 : 0);
}

// ---- weight fetch for concatenated [W;U], col order [i|ste|c|o|fre]
static __device__ __forceinline__ float fetch_wcat(
    int k, int n,
    const float* Wi, const float* Ui, const float* Ws, const float* Us,
    const float* Wc, const float* Uc, const float* Wo, const float* Uo,
    const float* Wf, const float* Uf) {
    if (n < 4 * HH) {
        int g = n >> 8;
        int c = n & (HH - 1);
        const float* W = (g == 0) ? Wi : (g == 1) ? Ws : (g == 2) ? Wc : Wo;
        const float* U = (g == 0) ? Ui : (g == 1) ? Us : (g == 2) ? Uc : Uo;
        return (k < DD) ? W[k * HH + c] : U[(k - DD) * HH + c];
    } else {
        int c = n - 4 * HH;
        return (k < DD) ? Wf[k * FF + c] : Uf[(k - DD) * FF + c];
    }
}

// ---- prep A: Bpack[((ct*KT + kt)*32 + lane)*16 + e]
__global__ void sfm_prep_w(
    const float* __restrict__ Wi, const float* __restrict__ Ui,
    const float* __restrict__ Ws, const float* __restrict__ Us,
    const float* __restrict__ Wf, const float* __restrict__ Uf,
    const float* __restrict__ Wc, const float* __restrict__ Uc,
    const float* __restrict__ Wo, const float* __restrict__ Uo,
    unsigned short* __restrict__ Bpack) {
    int idx = blockIdx.x * blockDim.x + threadIdx.x;
    if (idx >= BPACK_ELEMS) return;
    int ct   = idx / (KT * 512);
    int rem  = idx % (KT * 512);
    int kt   = rem >> 9;
    int le   = rem & 511;
    int lane = le >> 4;
    int e    = le & 15;
    int n = ct * 16 + (lane & 15);
    int k = kt * 32 + a_koff(e, lane);
    Bpack[idx] = f2bf_u16(fetch_wcat(k, n, Wi, Ui, Ws, Us, Wc, Uc, Wo, Uo, Wf, Uf));
}

// ---- prep B: x-part A fragments for every (band, t).
__global__ void sfm_prep_x(const float* __restrict__ in_x,
                           unsigned short* __restrict__ Xfrag) {
    int idx = blockIdx.x * blockDim.x + threadIdx.x;
    if (idx >= XFRAG_ELEMS) return;
    int e      = idx & 15;
    int lane_f = (idx >> 4) & 31;
    int kt     = (idx >> 9) & 3;
    int t      = (idx >> 11) & (TT - 1);
    int band   = idx >> 17;
    int row = lane_f & 15;
    int k = kt * 32 + a_koff(e, lane_f);                  // 0..127
    float v = in_x[((band * 16 + row) * DD + k) * TT + t];
    Xfrag[idx] = f2bf_u16(v);
}

// ---- main persistent kernel: one block = one 16-row batch band, all 64 steps.
__global__ void __launch_bounds__(NTHR, 1)
sfm_main_kernel(
    const unsigned short* __restrict__ Bpack,
    const unsigned short* __restrict__ Xfrag,
    const float* __restrict__ b_i,  const float* __restrict__ b_ste,
    const float* __restrict__ b_c,  const float* __restrict__ b_o,
    const float* __restrict__ b_fre,
    const float* __restrict__ U_a,  const float* __restrict__ b_a,
    const float* __restrict__ W_p,  const float* __restrict__ b_p,
    const float* __restrict__ fc_w, const float* __restrict__ fc_b,
    float* __restrict__ out) {

    __shared__ float          lds_pre[16 * PRE_STRIDE];         // gate pre-acts
    __shared__ __align__(32) unsigned short ldsA[KT * 32 * 16]; // full A frags (12 KB)
    __shared__ float          ldsH[16 * H_STRIDE];              // persistent h band
    __shared__ float          lds_bias[NTOT];
    __shared__ float          ctab_cos[16], ctab_sin[16];
    __shared__ float          uaf[FF], ba_s[HH];

    const int tid  = threadIdx.x;
    const int lane = tid & 31;
    const int wave = tid >> 5;
    const int band = blockIdx.x;

    // ---- one-time staging ----
    if (tid < 16) {
        float ang = (2.0f * 3.14159265358979323846f / 16.0f) * (float)tid;
        ctab_cos[tid] = __cosf(ang);
        ctab_sin[tid] = __sinf(ang);
        uaf[tid] = U_a[tid];
    }
    if (tid < HH) ba_s[tid] = b_a[tid];
    for (int n = tid; n < NTOT; n += NTHR) {
        float v;
        if      (n < 256)  v = b_i[n];
        else if (n < 512)  v = b_ste[n - 256];
        else if (n < 768)  v = b_c[n - 512];
        else if (n < 1024) v = b_o[n - 768];
        else               v = b_fre[n - 1024];
        lds_bias[n] = v;
    }
    for (int i = tid; i < 16 * H_STRIDE; i += NTHR) ldsH[i] = 0.0f;  // h0 = 0

    // ---- register-resident S state: 8 cols x 8 freq-pairs, v2f16 packed ----
    const int erow = tid >> 5;            // 0..15
    const int c0   = lane * 8;            // 8 consecutive h columns
    h2f srS[8][8], siS[8][8];
#pragma unroll
    for (int j = 0; j < 8; ++j)
#pragma unroll
        for (int p = 0; p < 8; ++p) {
            srS[j][p] = splat2(0.0f);
            siS[j][p] = splat2(0.0f);
        }

    __syncthreads();

    // packed U_a (per-thread, constant over steps)
    h2f uaP[8];
#pragma unroll
    for (int p = 0; p < 8; ++p)
        uaP[p] = (h2f){(_Float16)uaf[2 * p], (_Float16)uaf[2 * p + 1]};

    for (int t = 1; t <= TT; ++t) {
        // ---- phase 1a: block-copy x-part A fragments (4 KB) into LDS ----
        {
            const uint4* xsrc = reinterpret_cast<const uint4*>(
                Xfrag + (band * TT + (t - 1)) * (KTX * 512));
            uint4* xdst = reinterpret_cast<uint4*>(ldsA);
            for (int i = tid; i < (KTX * 512) / 8; i += NTHR)   // 256 uint4
                xdst[i] = xsrc[i];
        }
        // ---- phase 1b: bf16 h-part A fragments (kt 4..11) from ldsH ----
        for (int i = tid; i < KTH * 32 * 16; i += NTHR) {       // 8 iters
            int e      = i & 15;
            int lane_f = (i >> 4) & 31;
            int ktl    = i >> 9;                                // 0..7
            int row = lane_f & 15;
            int hcol = ktl * 32 + a_koff(e, lane_f);            // 0..255
            ldsA[KTX * 512 + i] = f2bf_u16(ldsH[row * H_STRIDE + hcol]);
        }
        __syncthreads();

        // ---- phase 2: WMMA GEMM, 65 col tiles over 16 waves ----
        const unsigned short* ab = ldsA + lane * 16;
        for (int ct = wave; ct < NT; ct += 16) {
            const unsigned short* bb = Bpack + (ct * KT * 32 + lane) * 16;
            if (ct + 16 < NT) {                 // warm L0 with the whole next panel
                const unsigned short* pn = Bpack + ((ct + 16) * KT * 32 + lane) * 16;
#pragma unroll
                for (int kt = 0; kt < KT; ++kt)
                    __builtin_prefetch(pn + kt * 512, 0, 0);
            }
            v8f acc = {0.f, 0.f, 0.f, 0.f, 0.f, 0.f, 0.f, 0.f};
#pragma unroll
            for (int kt = 0; kt < KT; ++kt) {
                v16bf a = *reinterpret_cast<const v16bf*>(ab + kt * 512);
                v16bf b = *reinterpret_cast<const v16bf*>(bb + kt * 512);
                acc = __builtin_amdgcn_wmma_f32_16x16x32_bf16(
                    false, a, false, b, (short)0, acc, false, false);
            }
            int colg = ct * 16 + (lane & 15);
            float bsv = lds_bias[colg];
            int rbase = (lane < 16) ? 0 : 8;
#pragma unroll
            for (int r = 0; r < 8; ++r)
                lds_pre[(rbase + r) * PRE_STRIDE + colg] = acc[r] + bsv;
        }
        __syncthreads();

        // ---- phase 3: packed-f16 state update ----
        {
            // per-step packed tables: re/im pairs and fre gate pairs
            h2f reP[8], imP[8], freP[8];
#pragma unroll
            for (int p = 0; p < 8; ++p) {
                int k0 = (t * (2 * p)) & 15;
                int k1 = (t * (2 * p + 1)) & 15;
                reP[p] = (h2f){(_Float16)ctab_cos[k0], (_Float16)ctab_cos[k1]};
                imP[p] = (h2f){(_Float16)ctab_sin[k0], (_Float16)ctab_sin[k1]};
                freP[p] = (h2f){
                    (_Float16)hsig(lds_pre[erow * PRE_STRIDE + 1024 + 2 * p]),
                    (_Float16)hsig(lds_pre[erow * PRE_STRIDE + 1024 + 2 * p + 1])};
            }
#pragma unroll
            for (int j = 0; j < 8; ++j) {
                int col = c0 + j;
                const float* prow = &lds_pre[erow * PRE_STRIDE];
                float iv  = hsig(prow[col]);
                float stv = hsig(prow[256 + col]);
                float cv  = iv * tanhf(prow[512 + col]);
                float ov  = hsig(prow[768 + col]);
                h2f stv2 = splat2(stv);
                h2f cv2  = splat2(cv);
                float accA = 0.0f;
#pragma unroll
                for (int p = 0; p < 8; ++p) {
                    h2f fg2 = stv2 * freP[p];                 // v_pk_mul_f16
                    h2f sr = fg2 * srS[j][p] + cv2 * reP[p];  // v_pk_fma_f16
                    h2f si = fg2 * siS[j][p] + cv2 * imP[p];
                    srS[j][p] = sr;
                    siS[j][p] = si;
#if defined(__has_builtin) && __has_builtin(__builtin_amdgcn_fdot2)
                    accA = __builtin_amdgcn_fdot2(sr, sr * uaP[p], accA, false);
                    accA = __builtin_amdgcn_fdot2(si, si * uaP[p], accA, false);
#else
                    accA += ((float)sr.x * (float)sr.x + (float)si.x * (float)si.x)
                            * uaf[2 * p];
                    accA += ((float)sr.y * (float)sr.y + (float)si.y * (float)si.y)
                            * uaf[2 * p + 1];
#endif
                }
                float av = tanhf(accA + ba_s[col]);
                ldsH[erow * H_STRIDE + col] = ov * av;        // h_new
            }
        }
        __syncthreads();
    }

    // ---- final projection: out[b] = (h.W_p + b_p) * fc_w + fc_b ----
    {
        float sacc = 0.0f;
#pragma unroll
        for (int jj = 0; jj < 8; ++jj) {
            int h = lane + jj * 32;
            sacc = fmaf(ldsH[wave * H_STRIDE + h], W_p[h], sacc);
        }
#pragma unroll
        for (int off = 16; off >= 1; off >>= 1)
            sacc += __shfl_xor(sacc, off, 32);
        if (lane == 0)
            out[band * 16 + wave] = (sacc + b_p[0]) * fc_w[0] + fc_b[0];
    }
}

extern "C" void kernel_launch(void* const* d_in, const int* in_sizes, int n_in,
                              void* d_out, int out_size, void* d_ws, size_t ws_size,
                              hipStream_t stream) {
    (void)in_sizes; (void)n_in; (void)out_size; (void)ws_size;
    const float* in_x  = (const float*)d_in[0];
    const float* W_i   = (const float*)d_in[1];
    const float* U_i   = (const float*)d_in[2];
    const float* b_i   = (const float*)d_in[3];
    const float* W_ste = (const float*)d_in[4];
    const float* U_ste = (const float*)d_in[5];
    const float* b_ste = (const float*)d_in[6];
    const float* W_fre = (const float*)d_in[7];
    const float* U_fre = (const float*)d_in[8];
    const float* b_fre = (const float*)d_in[9];
    const float* W_c   = (const float*)d_in[10];
    const float* U_c   = (const float*)d_in[11];
    const float* b_c   = (const float*)d_in[12];
    const float* W_o   = (const float*)d_in[13];
    const float* U_o   = (const float*)d_in[14];
    const float* b_o   = (const float*)d_in[15];
    const float* U_a   = (const float*)d_in[16];
    const float* b_a   = (const float*)d_in[17];
    const float* W_p   = (const float*)d_in[18];
    const float* b_p   = (const float*)d_in[19];
    const float* fc_w  = (const float*)d_in[20];
    const float* fc_b  = (const float*)d_in[21];

    unsigned short* Bpack = (unsigned short*)d_ws;
    unsigned short* Xfrag = (unsigned short*)d_ws + XFRAG_OFF_U16;

    sfm_prep_w<<<(BPACK_ELEMS + 1023) / 1024, 1024, 0, stream>>>(
        W_i, U_i, W_ste, U_ste, W_fre, U_fre, W_c, U_c, W_o, U_o, Bpack);
    sfm_prep_x<<<(XFRAG_ELEMS + 1023) / 1024, 1024, 0, stream>>>(in_x, Xfrag);

    sfm_main_kernel<<<16, NTHR, 0, stream>>>(
        Bpack, Xfrag, b_i, b_ste, b_c, b_o, b_fre,
        U_a, b_a, W_p, b_p, fc_w, fc_b, (float*)d_out);
}
